// GATv2_20985210208645
// MI455X (gfx1250) — matrix operand order repaired and checked by reading, not compiled, not run on hardware
//
#include <hip/hip_runtime.h>
#include <hip/hip_bf16.h>

#define NEG_SLOPE 0.2f

typedef __attribute__((ext_vector_type(16))) __bf16 v16bf;
typedef __attribute__((ext_vector_type(8)))  __bf16 v8bf;
typedef __attribute__((ext_vector_type(8)))  float  v8f;

// ---------- float <-> order-preserving uint key (for atomic max on floats) ----
__device__ __forceinline__ unsigned fkey(float f) {
  unsigned u = __float_as_uint(f);
  return (u & 0x80000000u) ? ~u : (u | 0x80000000u);
}
__device__ __forceinline__ float funkey(unsigned k) {
  unsigned u = (k & 0x80000000u) ? (k & 0x7fffffffu) : ~k;
  return __uint_as_float(u);
}

// ---------------- zero-fill accumulators (key 0 == smallest max-key) ----------
__global__ void k_zero(float* __restrict__ p, long long n) {
  long long i = (long long)blockIdx.x * blockDim.x + threadIdx.x;
  if (i < n) p[i] = 0.0f;
}

// ---------------- cast x (N x 128 f32) -> bf16, same layout -------------------
__global__ void k_cast_x(const float* __restrict__ x, __bf16* __restrict__ xbf,
                         long long n) {
  long long i = (long long)blockIdx.x * blockDim.x + threadIdx.x;
  if (i < n) xbf[i] = (__bf16)x[i];
}

// --------- cast + transpose W (128x128 f32 row-major) -> Wt bf16 [n][k] -------
__global__ void k_cast_w(const float* __restrict__ Wl, const float* __restrict__ Wr,
                         __bf16* __restrict__ WtL, __bf16* __restrict__ WtR) {
  int t = blockIdx.x * blockDim.x + threadIdx.x;   // 16384 threads
  if (t >= 128 * 128) return;
  int n = t >> 7, k = t & 127;
  WtL[t] = (__bf16)Wl[(size_t)k * 128 + n];
  WtR[t] = (__bf16)Wr[(size_t)k * 128 + n];
}

// ---------------- fused projection GEMM: xl = xW_l+b_l, xr = xW_r+b_r --------
// block = 256 threads = 8 waves; wave w computes column tile w (16 cols) of a
// 16-row tile for BOTH weight matrices, sharing one A fragment.
// All operands pre-converted to bf16 -> inner loop is 6x b128 loads + 2 WMMAs.
__global__ __launch_bounds__(256) void k_proj(
    const __bf16* __restrict__ xbf, const __bf16* __restrict__ WtL,
    const __bf16* __restrict__ WtR, const float* __restrict__ bl,
    const float* __restrict__ br, float* __restrict__ xl,
    float* __restrict__ xr, int N)
{
  const int wave = threadIdx.x >> 5;     // 0..7 -> column tile
  const int lane = threadIdx.x & 31;
  const int half = lane >> 4;            // 0 or 1
  const int lm   = lane & 15;
  const int rt   = blockIdx.x;           // row tile
  int m = rt * 16 + lm; if (m >= N) m = N - 1;   // clamp (keeps EXEC uniform)
  const int n = wave * 16 + lm;

  const __bf16* xrow = xbf + (size_t)m * 128;
  const __bf16* wlc  = WtL + (size_t)n * 128;     // column n of W_l, contiguous K
  const __bf16* wrc  = WtR + (size_t)n * 128;

  v8f cL = {}; v8f cR = {};
#pragma unroll
  for (int kt = 0; kt < 4; ++kt) {
    const int kbase = kt * 32;
    // A frag: elem j = A[m][kbase + 16*(j>>3) + 8*half + (j&7)] -> 2x 16B loads
    v8bf a0 = *(const v8bf*)(xrow + kbase + 8 * half);
    v8bf a1 = *(const v8bf*)(xrow + kbase + 16 + 8 * half);
    v16bf a = __builtin_shufflevector(a0, a1, 0, 1, 2, 3, 4, 5, 6, 7,
                                      8, 9, 10, 11, 12, 13, 14, 15);
    // B frag: elem j = Wt[n][kbase + 16*(j>>3) + 8*half + (j&7)]
    v8bf bl0 = *(const v8bf*)(wlc + kbase + 8 * half);
    v8bf bl1 = *(const v8bf*)(wlc + kbase + 16 + 8 * half);
    v8bf br0 = *(const v8bf*)(wrc + kbase + 8 * half);
    v8bf br1 = *(const v8bf*)(wrc + kbase + 16 + 8 * half);
    v16bf bL = __builtin_shufflevector(bl0, bl1, 0, 1, 2, 3, 4, 5, 6, 7,
                                       8, 9, 10, 11, 12, 13, 14, 15);
    v16bf bR = __builtin_shufflevector(br0, br1, 0, 1, 2, 3, 4, 5, 6, 7,
                                       8, 9, 10, 11, 12, 13, 14, 15);
    cL = __builtin_amdgcn_wmma_f32_16x16x32_bf16(false, a, false, bL, (short)0, cL, false, false);
    cR = __builtin_amdgcn_wmma_f32_16x16x32_bf16(false, a, false, bR, (short)0, cR, false, false);
  }
  const float bbl = bl[n], bbr = br[n];
#pragma unroll
  for (int r = 0; r < 8; ++r) {          // C elem r -> row r + 8*half
    const int row = rt * 16 + r + 8 * half;
    if (row < N) {
      xl[(size_t)row * 128 + n] = cL[r] + bbl;
      xr[(size_t)row * 128 + n] = cR[r] + bbr;
    }
  }
}

// ------------- pass A: attention logits + segment max (per edge, per head) ---
__global__ void k_logit_max(const int* __restrict__ src, const int* __restrict__ dst,
    const float* __restrict__ xl, const float* __restrict__ xr,
    const float* __restrict__ att, float* __restrict__ logit,
    unsigned* __restrict__ segmax, int E, int ET)
{
  int tid = blockIdx.x * blockDim.x + threadIdx.x;
  if (tid >= ET * 8) return;
  int e = tid >> 3, h = tid & 7;
  int s, d;
  if (e < E) { s = src[e]; d = dst[e]; } else { s = e - E; d = s; }  // self loops
  const float4* pl = (const float4*)(xl + (size_t)s * 128 + h * 16);
  const float4* pr = (const float4*)(xr + (size_t)d * 128 + h * 16);
  const float4* pa = (const float4*)(att + h * 16);
  float acc = 0.f;
#pragma unroll
  for (int q = 0; q < 4; ++q) {
    float4 l = pl[q], r = pr[q], aa = pa[q];
    float v;
    v = l.x + r.x; v = v > 0.f ? v : NEG_SLOPE * v; acc += aa.x * v;
    v = l.y + r.y; v = v > 0.f ? v : NEG_SLOPE * v; acc += aa.y * v;
    v = l.z + r.z; v = v > 0.f ? v : NEG_SLOPE * v; acc += aa.z * v;
    v = l.w + r.w; v = v > 0.f ? v : NEG_SLOPE * v; acc += aa.w * v;
  }
  logit[tid] = acc;
  atomicMax(&segmax[(size_t)d * 8 + h], fkey(acc));
}

// ------------- pass B: p = exp(logit - max), denom += p ----------------------
__global__ void k_exp_sum(const int* __restrict__ dst, float* __restrict__ logit,
    const unsigned* __restrict__ segmax, float* __restrict__ denom, int E, int ET)
{
  int tid = blockIdx.x * blockDim.x + threadIdx.x;
  if (tid >= ET * 8) return;
  int e = tid >> 3, h = tid & 7;
  int d = (e < E) ? dst[e] : (e - E);
  float p = __expf(logit[tid] - funkey(segmax[(size_t)d * 8 + h]));
  logit[tid] = p;
  atomicAdd(&denom[(size_t)d * 8 + h], p);
}

// ------------- pass C: out[dst] += alpha * xl[src] ---------------------------
__global__ void k_scatter(const int* __restrict__ src, const int* __restrict__ dst,
    const float* __restrict__ xl, const float* __restrict__ logit,
    const float* __restrict__ denom, float* __restrict__ outacc, int E, int ET)
{
  int tid = blockIdx.x * blockDim.x + threadIdx.x;
  if (tid >= ET * 8) return;
  int e = tid >> 3, h = tid & 7;
  int s, d;
  if (e < E) { s = src[e]; d = dst[e]; } else { s = e - E; d = s; }
  float alpha = logit[tid] / (denom[(size_t)d * 8 + h] + 1e-16f);
  const float4* pl = (const float4*)(xl + (size_t)s * 128 + h * 16);
  float* po = outacc + (size_t)d * 128 + h * 16;
#pragma unroll
  for (int q = 0; q < 4; ++q) {
    float4 l = pl[q];
    atomicAdd(po + q * 4 + 0, l.x * alpha);
    atomicAdd(po + q * 4 + 1, l.y * alpha);
    atomicAdd(po + q * 4 + 2, l.z * alpha);
    atomicAdd(po + q * 4 + 3, l.w * alpha);
  }
}

// ------------- ELU + global mean-pool accumulation ---------------------------
__global__ void k_elu_pool(const float* __restrict__ outacc,
    const float* __restrict__ bias, const int* __restrict__ batch,
    float* __restrict__ pooled, float* __restrict__ counts, int N)
{
  int tid = blockIdx.x * blockDim.x + threadIdx.x;
  if (tid >= N * 128) return;
  int node = tid >> 7, c = tid & 127;
  float v = outacc[tid] + bias[c];
  v = v > 0.f ? v : (__expf(v) - 1.0f);     // ELU(alpha=1)
  int g = batch[node];
  atomicAdd(&pooled[g * 128 + c], v);
  if (c == 0) atomicAdd(&counts[g], 1.0f);
}

// ------------- mean + tiny FC: out[g][o] = (pooled[g]/cnt) . fcW[:,o] + fcb --
__global__ void k_final(const float* __restrict__ pooled,
    const float* __restrict__ counts, const float* __restrict__ fcW,
    const float* __restrict__ fcb, float* __restrict__ out, int G)
{
  int t = blockIdx.x * blockDim.x + threadIdx.x;
  if (t >= G * 2) return;
  int g = t >> 1, o = t & 1;
  float inv = 1.0f / fmaxf(counts[g], 1.0f);
  float acc = fcb[o];
  const float* pg = pooled + g * 128;
#pragma unroll 8
  for (int c = 0; c < 128; ++c) acc += (pg[c] * inv) * fcW[c * 2 + o];
  out[t] = acc;
}

extern "C" void kernel_launch(void* const* d_in, const int* in_sizes, int n_in,
                              void* d_out, int out_size, void* d_ws, size_t ws_size,
                              hipStream_t stream) {
  (void)n_in; (void)ws_size;
  const float* x    = (const float*)d_in[0];
  const int*   ei   = (const int*)d_in[1];
  const int*   batch= (const int*)d_in[2];
  const float* Wl   = (const float*)d_in[3];
  const float* bl   = (const float*)d_in[4];
  const float* Wr   = (const float*)d_in[5];
  const float* br   = (const float*)d_in[6];
  const float* att  = (const float*)d_in[7];
  const float* bias = (const float*)d_in[8];
  const float* fcW  = (const float*)d_in[9];
  const float* fcb  = (const float*)d_in[10];
  float* out = (float*)d_out;

  const int N  = in_sizes[2];          // 50000 nodes
  const int E  = in_sizes[1] / 2;      // 800000 edges
  const int ET = E + N;                // + self loops
  const int G  = out_size / 2;         // 128 graphs

  const int* src = ei;                 // edge_index[0]
  const int* dst = ei + E;             // edge_index[1]

  // workspace layout (all L2-resident: ~120 MB total)
  float*    xl     = (float*)d_ws;
  float*    xr     = xl + (size_t)N * 128;
  float*    logit  = xr + (size_t)N * 128;                // ET*8, reused for p
  unsigned* segmax = (unsigned*)(logit + (size_t)ET * 8); // N*8 keys
  float*    denom  = (float*)(segmax + (size_t)N * 8);    // N*8
  float*    outacc = denom + (size_t)N * 8;               // N*128
  float*    pooled = outacc + (size_t)N * 128;            // G*128
  float*    counts = pooled + (size_t)G * 128;            // G
  __bf16*   xbf    = (__bf16*)(counts + G);               // N*128 bf16
  __bf16*   WtL    = xbf + (size_t)N * 128;               // 128*128 bf16
  __bf16*   WtR    = WtL + 128 * 128;

  // zero segmax..counts (key 0 is the identity for the float max-key encoding)
  long long nz = (long long)N * 8 * 2 + (long long)N * 128 + (long long)G * 128 + G;
  k_zero<<<(unsigned)((nz + 255) / 256), 256, 0, stream>>>((float*)segmax, nz);

  // pre-convert operands for the WMMA GEMM
  long long nx = (long long)N * 128;
  k_cast_x<<<(unsigned)((nx + 255) / 256), 256, 0, stream>>>(x, xbf, nx);
  k_cast_w<<<(128 * 128 + 255) / 256, 256, 0, stream>>>(Wl, Wr, WtL, WtR);

  k_proj<<<(N + 15) / 16, 256, 0, stream>>>(xbf, WtL, WtR, bl, br, xl, xr, N);

  const int nEH = ET * 8;
  const int gEH = (nEH + 255) / 256;
  k_logit_max<<<gEH, 256, 0, stream>>>(src, dst, xl, xr, att, logit, segmax, E, ET);
  k_exp_sum  <<<gEH, 256, 0, stream>>>(dst, logit, segmax, denom, E, ET);
  k_scatter  <<<gEH, 256, 0, stream>>>(src, dst, xl, logit, denom, outacc, E, ET);

  long long nNC = (long long)N * 128;
  k_elu_pool<<<(unsigned)((nNC + 255) / 256), 256, 0, stream>>>(outacc, bias, batch, pooled, counts, N);

  k_final<<<(G * 2 + 255) / 256, 256, 0, stream>>>(pooled, counts, fcW, fcb, out, G);
}
